// Qwen3MoeDecoderLayer_90134183674317
// MI455X (gfx1250) — matrix operand order repaired and checked by reading, not compiled, not run on hardware
//
#include <hip/hip_runtime.h>
#include <hip/hip_bf16.h>
#include <math.h>

typedef __bf16 bf16;
typedef __attribute__((ext_vector_type(16))) __bf16 v16b;
typedef __attribute__((ext_vector_type(8)))  __bf16 v8b;
typedef __attribute__((ext_vector_type(8)))  float  v8f;

#define T_    1024
#define H_    1024
#define NH_   8
#define NKV_  4
#define D_    128
#define E_    32
#define KSEL_ 4
#define I_    512
#define QKVW_ ((NH_ + 2 * NKV_) * D_)   // 2048

// GEMM tiling: block = 4 waves stacked along M. Block tile 128x64, wave tile 32x64.
#define BM_ 128
#define WM_ 32
#define BN_ 64
#define LDSK_ 40   // padded inner dim (32 K elems + 8 pad) to spread LDS banks

// ---------------------------------------------------------------------------
// WMMA fragment helpers (wave32; A/B use contiguous k order, C/D per CDNA5 ISA 7.12.2)
// ---------------------------------------------------------------------------

// A fragment: 16x32 bf16 tile, row-major source, K contiguous.
// lane L<16: row M=L, K = k0+0..15 (elems 0..15)
// lane L>=16: row M=L-16, K = k0+16..31
__device__ __forceinline__ v16b load_a_frag_ptr(const bf16* rowp, int k0) {
  int lane  = threadIdx.x & 31;
  int kbase = k0 + ((lane >> 4) << 4);
  const bf16* p = rowp + kbase;
  v8b lo = *(const v8b*)(p);
  v8b hi = *(const v8b*)(p + 8);
  return __builtin_shufflevector(lo, hi, 0,1,2,3,4,5,6,7,8,9,10,11,12,13,14,15);
}

__device__ __forceinline__ v16b load_a_frag(const bf16* A, int lda, int m0, int k0) {
  int lane = threadIdx.x & 31;
  int row  = m0 + (lane & 15);
  return load_a_frag_ptr(A + (size_t)row * lda, k0);
}

// B fragment from global: W row-major (N x K). lane L<16: col n0+L, K k0..k0+15;
// lane L>=16: col n0+L-16, K k0+16..k0+31
__device__ __forceinline__ v16b load_b_frag(const bf16* W, int ldb, int n0, int k0) {
  int lane  = threadIdx.x & 31;
  int col   = n0 + (lane & 15);
  int kbase = k0 + ((lane >> 4) << 4);
  const bf16* p = W + (size_t)col * ldb + kbase;
  v8b lo = *(const v8b*)(p);
  v8b hi = *(const v8b*)(p + 8);
  return __builtin_shufflevector(lo, hi, 0,1,2,3,4,5,6,7,8,9,10,11,12,13,14,15);
}

// B fragment from an LDS-staged tile laid out [64 cols][LDSK_], K contiguous.
__device__ __forceinline__ v16b load_b_frag_lds(const bf16* tile, int subn) {
  int lane  = threadIdx.x & 31;
  int col   = subn * 16 + (lane & 15);
  int kbase = (lane >> 4) << 4;
  const bf16* p = tile + col * LDSK_ + kbase;
  v8b lo = *(const v8b*)(p);
  v8b hi = *(const v8b*)(p + 8);
  return __builtin_shufflevector(lo, hi, 0,1,2,3,4,5,6,7,8,9,10,11,12,13,14,15);
}

// C/D fragment: VGPR r -> M = m0 + (lane<16 ? r : 8+r), N = n0 + (lane&15)
__device__ __forceinline__ void store_c_f32(float* C, int ldc, int m0, int n0,
                                            const v8f& c, int M, int N) {
  int lane  = threadIdx.x & 31;
  int n     = n0 + (lane & 15);
  int mbase = m0 + ((lane >> 4) << 3);
  if (n >= N) return;
#pragma unroll
  for (int r = 0; r < 8; ++r) {
    int m = mbase + r;
    if (m < M) C[(size_t)m * ldc + n] = c[r];
  }
}

__device__ __forceinline__ v8f wmma_bf16(const v16b& a, const v16b& b, const v8f& c) {
  return __builtin_amdgcn_wmma_f32_16x16x32_bf16(false, a, false, b, (short)0, c,
                                                 false, false);
}

// One-hop global -> LDS async copy (gfx1250, ASYNCcnt-tracked).
// Generic LDS pointer's low 32 bits are the LDS byte offset (aperture mapping).
__device__ __forceinline__ void async_copy_b128(const bf16* gsrc, bf16* ldst) {
  unsigned loff = (unsigned)(unsigned long long)ldst;
  unsigned long long gaddr = (unsigned long long)gsrc;
  asm volatile("global_load_async_to_lds_b128 %0, %1, off"
               :: "v"(loff), "v"(gaddr) : "memory");
}

__device__ __forceinline__ void wait_async0() {
  asm volatile("s_wait_asynccnt 0x0" ::: "memory");
}

// ---------------------------------------------------------------------------
// Generic NT GEMM: C(MxN,f32) = A(MxK,bf16) * W(NxK,bf16)^T
// Block tile 128x64 (4 waves x 32 rows), wave tile 32x64 (2x4 accumulators).
// B tile (64 cols x 32 K) is async-staged into LDS once per K-step, double
// buffered, and shared by all 4 waves.
// mode: 0 = plain, 1 = causal block skip (scores), 2 = K clamp to diag (P*V)
// ---------------------------------------------------------------------------
__global__ __launch_bounds__(128) void gemm_nt(
    const bf16* __restrict__ A, const bf16* __restrict__ W, float* __restrict__ C,
    int M, int N, int K, int lda, int ldb, int ldc, int mode) {
  __shared__ bf16 ldsB[2][BN_][LDSK_];
  int wave = threadIdx.x >> 5;
  int n0   = blockIdx.x * BN_;
  int mblk = blockIdx.y * BM_;
  int m0w  = mblk + wave * WM_;
  if (mode == 1 && n0 > mblk + BM_ - 1) return;   // fully-masked causal block
  int kmax = K;
  if (mode == 2) kmax = min(K, mblk + BM_);       // probs zero for s > t
  int nsteps = (kmax + 31) >> 5;

  // stage B tile for K-step j into buffer buf: 256 16-byte chunks / 128 threads
  auto stage = [&](int buf, int k0) {
#pragma unroll
    for (int it = 0; it < 2; ++it) {
      int chunk = threadIdx.x + it * 128;       // 0..255
      int col   = chunk >> 2;                   // 0..63
      int kk    = (chunk & 3) * 8;              // 0,8,16,24
      int gcol  = min(n0 + col, N - 1);
      async_copy_b128(W + (size_t)gcol * ldb + k0 + kk, &ldsB[buf][col][kk]);
    }
  };

  stage(0, 0);
  v8f acc[2][4] = {};
  for (int j = 0; j < nsteps; ++j) {
    wait_async0();        // my staged chunks have landed
    __syncthreads();      // everyone's chunks have landed; prev buffer free
    if (j + 1 < nsteps) stage((j + 1) & 1, (j + 1) << 5);
    int k0 = j << 5;
    v16b a0 = load_a_frag(A, lda, m0w, k0);
    v16b a1 = load_a_frag(A, lda, m0w + 16, k0);
    const bf16* tile = &ldsB[j & 1][0][0];
#pragma unroll
    for (int sn = 0; sn < 4; ++sn) {
      v16b b = load_b_frag_lds(tile, sn);
      acc[0][sn] = wmma_bf16(a0, b, acc[0][sn]);
      acc[1][sn] = wmma_bf16(a1, b, acc[1][sn]);
    }
  }
#pragma unroll
  for (int sm = 0; sm < 2; ++sm)
#pragma unroll
    for (int sn = 0; sn < 4; ++sn)
      store_c_f32(C, ldc, m0w + sm * 16, n0 + sn * 16, acc[sm][sn], M, N);
}

// ---------------------------------------------------------------------------
// Block reductions (blockDim multiple of 32, <= 256)
// ---------------------------------------------------------------------------
__device__ __forceinline__ float block_reduce_sum(float v, float* sh) {
  int lane = threadIdx.x & 31, w = threadIdx.x >> 5, nw = blockDim.x >> 5;
#pragma unroll
  for (int o = 16; o > 0; o >>= 1) v += __shfl_xor(v, o, 32);
  if (lane == 0) sh[w] = v;
  __syncthreads();
  float s = (threadIdx.x < nw) ? sh[threadIdx.x] : 0.f;
  if (w == 0) {
#pragma unroll
    for (int o = 4; o > 0; o >>= 1) s += __shfl_xor(s, o, 32);
    if (lane == 0) sh[0] = s;
  }
  __syncthreads();
  float r = sh[0];
  __syncthreads();
  return r;
}

__device__ __forceinline__ float block_reduce_max(float v, float* sh) {
  int lane = threadIdx.x & 31, w = threadIdx.x >> 5, nw = blockDim.x >> 5;
#pragma unroll
  for (int o = 16; o > 0; o >>= 1) v = fmaxf(v, __shfl_xor(v, o, 32));
  if (lane == 0) sh[w] = v;
  __syncthreads();
  float s = (threadIdx.x < nw) ? sh[threadIdx.x] : -3.4e38f;
  if (w == 0) {
#pragma unroll
    for (int o = 4; o > 0; o >>= 1) s = fmaxf(s, __shfl_xor(s, o, 32));
    if (lane == 0) sh[0] = s;
  }
  __syncthreads();
  float r = sh[0];
  __syncthreads();
  return r;
}

// ---------------------------------------------------------------------------
// Weight conversion kernels
// ---------------------------------------------------------------------------
__global__ void cvt_bf16(const float* __restrict__ in, bf16* __restrict__ out, int n) {
  for (int i = blockIdx.x * blockDim.x + threadIdx.x; i < n; i += gridDim.x * blockDim.x)
    out[i] = (bf16)in[i];
}

// per-expert transpose-convert: z-th slab (R x Ccols) f32 -> (Ccols x R) bf16
__global__ void cvt_t_bf16(const float* __restrict__ in, bf16* __restrict__ out,
                           int R, int Ccols) {
  size_t slab = (size_t)R * Ccols;
  const float* inz = in + blockIdx.z * slab;
  bf16* outz = out + blockIdx.z * slab;
  for (size_t idx = blockIdx.x * blockDim.x + threadIdx.x; idx < slab;
       idx += (size_t)gridDim.x * blockDim.x) {
    int r = (int)(idx % R);
    int c = (int)(idx / R);
    outz[idx] = (bf16)inz[(size_t)r * Ccols + c];
  }
}

// ---------------------------------------------------------------------------
// Fused residual add + RMSNorm: res = a + b; h = rms(res) * w  (bf16)
// ---------------------------------------------------------------------------
__global__ void add_rmsnorm(const float* __restrict__ a, const float* __restrict__ b,
                            const float* __restrict__ w, float* __restrict__ res,
                            bf16* __restrict__ h) {
  __shared__ float sh[8];
  int t = blockIdx.x;
  const float* pa = a + (size_t)t * H_;
  const float* pb = b + (size_t)t * H_;
  float vals[4];
  float ss = 0.f;
#pragma unroll
  for (int i = 0; i < 4; ++i) {
    int idx = threadIdx.x + i * 256;
    float v = pa[idx] + pb[idx];
    vals[i] = v;
    ss += v * v;
  }
  ss = block_reduce_sum(ss, sh);
  float rinv = rsqrtf(ss * (1.0f / H_) + 1e-6f);
#pragma unroll
  for (int i = 0; i < 4; ++i) {
    int idx = threadIdx.x + i * 256;
    res[(size_t)t * H_ + idx] = vals[i];
    h[(size_t)t * H_ + idx] = (bf16)(vals[i] * rinv * w[idx]);
  }
}

// ---------------------------------------------------------------------------
// Q/K RMSNorm + RoPE + V transpose. One block (128 thr) per token.
// ---------------------------------------------------------------------------
__global__ void qk_rope(const float* __restrict__ qkv, const int* __restrict__ pos,
                        const float* __restrict__ qnw, const float* __restrict__ knw,
                        bf16* __restrict__ qo, bf16* __restrict__ ko,
                        bf16* __restrict__ vTo) {
  __shared__ float pair[128];
  __shared__ float red[8];
  int t = blockIdx.x;
  int d = threadIdx.x;
  const float* row = qkv + (size_t)t * QKVW_;
  float p = (float)pos[t];
  int j = d & 63;
  float freq = p * __powf(1.0e6f, -(float)j * (1.0f / 64.0f));
  float cs = cosf(freq), sn = sinf(freq);

  for (int h = 0; h < NH_ + NKV_; ++h) {
    bool isq = h < NH_;
    float x = row[h * D_ + d];
    float ss = block_reduce_sum(x * x, red);
    float rinv = rsqrtf(ss * (1.0f / D_) + 1e-6f);
    float xn = x * rinv * (isq ? qnw[d] : knw[d]);
    pair[d] = xn;
    __syncthreads();
    float other = (d < 64) ? pair[d + 64] : pair[d - 64];
    float out = (d < 64) ? (xn * cs - other * sn) : (xn * cs + other * sn);
    __syncthreads();
    if (isq) qo[((size_t)h * T_ + t) * D_ + d] = (bf16)out;
    else     ko[((size_t)(h - NH_) * T_ + t) * D_ + d] = (bf16)out;
  }
#pragma unroll
  for (int kh = 0; kh < NKV_; ++kh) {
    float v = row[(NH_ + NKV_ + kh) * D_ + d];
    vTo[((size_t)kh * D_ + d) * T_ + t] = (bf16)v;  // [NKV][D][T]
  }
}

// ---------------------------------------------------------------------------
// Causal softmax: probs = softmax(scale * scores[:, 0..t]), 0 beyond. grid (T,NH)
// ---------------------------------------------------------------------------
__global__ void softmax_causal(const float* __restrict__ scores, bf16* __restrict__ probs) {
  __shared__ float sh[8];
  int t = blockIdx.x, h = blockIdx.y;
  const float scale = 0.08838834764831845f;  // 128^-0.5
  const float* srow = scores + ((size_t)h * T_ + t) * T_;
  bf16* prow = probs + ((size_t)h * T_ + t) * T_;
  int n = t + 1;
  float mx = -3.4e38f;
  for (int i = threadIdx.x; i < n; i += 256) mx = fmaxf(mx, srow[i] * scale);
  mx = block_reduce_max(mx, sh);
  float sum = 0.f;
  for (int i = threadIdx.x; i < n; i += 256) sum += __expf(srow[i] * scale - mx);
  sum = block_reduce_sum(sum, sh);
  float inv = 1.0f / sum;
  for (int i = threadIdx.x; i < T_; i += 256) {
    float v = (i < n) ? __expf(srow[i] * scale - mx) * inv : 0.0f;
    prow[i] = (bf16)v;
  }
}

// ---------------------------------------------------------------------------
// Router: wave softmax over 32 logits, top-4, normalize, per-expert scatter.
// ---------------------------------------------------------------------------
__global__ void router_topk(const float* __restrict__ logits, int* __restrict__ cnt,
                            int* __restrict__ tok, float* __restrict__ slotw) {
  int t = blockIdx.x;
  int e = threadIdx.x;
  float lg = logits[(size_t)t * E_ + e];
  float mx = lg;
#pragma unroll
  for (int o = 16; o > 0; o >>= 1) mx = fmaxf(mx, __shfl_xor(mx, o, 32));
  float ex = __expf(lg - mx);
  float s = ex;
#pragma unroll
  for (int o = 16; o > 0; o >>= 1) s += __shfl_xor(s, o, 32);
  float rw = ex / s;

  float v = rw, wsum = 0.f, myw = 0.f;
  bool sel = false;
#pragma unroll
  for (int it = 0; it < KSEL_; ++it) {
    float m = v;
    int ml = e;
#pragma unroll
    for (int o = 16; o > 0; o >>= 1) {
      float ov = __shfl_xor(m, o, 32);
      int oi = __shfl_xor(ml, o, 32);
      if (ov > m || (ov == m && oi < ml)) { m = ov; ml = oi; }
    }
    wsum += m;
    if (e == ml) { sel = true; myw = v; v = -1.f; }
  }
  if (sel) {
    int slot = atomicAdd(&cnt[e], 1);
    tok[e * T_ + slot] = t;
    slotw[e * T_ + slot] = myw / wsum;
  }
}

// ---------------------------------------------------------------------------
// MoE gate+up with fused SiLU*up. grid (I/64, T/16, E), block 128 (4 waves).
// ---------------------------------------------------------------------------
__global__ void moe_gate_up(const bf16* __restrict__ h2, const bf16* __restrict__ wg_t,
                            const bf16* __restrict__ wu_t, const int* __restrict__ cnt,
                            const int* __restrict__ tok, bf16* __restrict__ act) {
  int e = blockIdx.z;
  int count = cnt[e];
  int m0 = blockIdx.y * 16;
  if (m0 >= count) return;
  int wave = threadIdx.x >> 5;
  int n0 = blockIdx.x * 64 + wave * 16;
  int lane = threadIdx.x & 31;
  int row = m0 + (lane & 15);
  int trow = tok[e * T_ + min(row, count - 1)];
  const bf16* arow = h2 + (size_t)trow * H_;
  const bf16* wge = wg_t + (size_t)e * I_ * H_;
  const bf16* wue = wu_t + (size_t)e * I_ * H_;
  v8f cg = {}, cu = {};
  for (int k0 = 0; k0 < H_; k0 += 32) {
    v16b a  = load_a_frag_ptr(arow, k0);
    v16b bg = load_b_frag(wge, H_, n0, k0);
    v16b bu = load_b_frag(wue, H_, n0, k0);
    cg = wmma_bf16(a, bg, cg);
    cu = wmma_bf16(a, bu, cu);
  }
  int n = n0 + (lane & 15);
  int mbase = m0 + ((lane >> 4) << 3);
#pragma unroll
  for (int r = 0; r < 8; ++r) {
    int m = mbase + r;
    if (m < count) {
      float g = cg[r], u = cu[r];
      float av = g / (1.0f + __expf(-g)) * u;      // silu(g) * u
      act[((size_t)e * T_ + m) * I_ + n] = (bf16)av;
    }
  }
}

// ---------------------------------------------------------------------------
// MoE down-proj + combine-weight scatter-add. grid (H/64, T/16, E), block 128.
// ---------------------------------------------------------------------------
__global__ void moe_down(const bf16* __restrict__ act, const bf16* __restrict__ wd_t,
                         const int* __restrict__ cnt, const int* __restrict__ tok,
                         const float* __restrict__ slotw, float* __restrict__ moe_out) {
  int e = blockIdx.z;
  int count = cnt[e];
  int m0 = blockIdx.y * 16;
  if (m0 >= count) return;
  int wave = threadIdx.x >> 5;
  int n0 = blockIdx.x * 64 + wave * 16;
  const bf16* ae = act + (size_t)e * T_ * I_;
  const bf16* wde = wd_t + (size_t)e * H_ * I_;
  v8f c = {};
  for (int k0 = 0; k0 < I_; k0 += 32) {
    v16b a = load_a_frag(ae, I_, m0, k0);    // junk rows beyond count masked at store
    v16b b = load_b_frag(wde, I_, n0, k0);
    c = wmma_bf16(a, b, c);
  }
  int lane = threadIdx.x & 31;
  int n = n0 + (lane & 15);
  int mbase = m0 + ((lane >> 4) << 3);
#pragma unroll
  for (int r = 0; r < 8; ++r) {
    int m = mbase + r;
    if (m < count) {
      int t = tok[e * T_ + m];
      float w = slotw[e * T_ + m];
      atomicAdd(&moe_out[(size_t)t * H_ + n], w * c[r]);
    }
  }
}

// ---------------------------------------------------------------------------
// Host launcher
// ---------------------------------------------------------------------------
extern "C" void kernel_launch(void* const* d_in, const int* in_sizes, int n_in,
                              void* d_out, int out_size, void* d_ws, size_t ws_size,
                              hipStream_t stream) {
  (void)in_sizes; (void)n_in; (void)out_size; (void)ws_size;
  const float* hidden   = (const float*)d_in[0];
  const float* residual = (const float*)d_in[1];
  const int*   positions= (const int*)  d_in[2];
  const float* qkv_w    = (const float*)d_in[3];
  const float* q_norm_w = (const float*)d_in[4];
  const float* k_norm_w = (const float*)d_in[5];
  const float* o_w      = (const float*)d_in[6];
  const float* ln1_w    = (const float*)d_in[7];
  const float* ln2_w    = (const float*)d_in[8];
  const float* gate_w   = (const float*)d_in[9];
  const float* w_gate   = (const float*)d_in[10];
  const float* w_up     = (const float*)d_in[11];
  const float* w_down   = (const float*)d_in[12];

  float* moe_out = (float*)d_out;                    // first T*H
  float* res_out = (float*)d_out + (size_t)T_ * H_;  // second T*H

  // ---- workspace layout ----
  char* base = (char*)d_ws;
  size_t off = 0;
  auto alloc = [&](size_t bytes) -> char* {
    char* p = base + off;
    off = (off + bytes + 255) & ~(size_t)255;
    return p;
  };
  bf16* qkvw_bf = (bf16*)alloc((size_t)QKVW_ * H_ * 2);
  bf16* ow_bf   = (bf16*)alloc((size_t)H_ * H_ * 2);
  bf16* gw_bf   = (bf16*)alloc((size_t)E_ * H_ * 2);
  bf16* wg_t    = (bf16*)alloc((size_t)E_ * I_ * H_ * 2);
  bf16* wu_t    = (bf16*)alloc((size_t)E_ * I_ * H_ * 2);
  bf16* wd_t    = (bf16*)alloc((size_t)E_ * H_ * I_ * 2);
  bf16* h_bf    = (bf16*)alloc((size_t)T_ * H_ * 2);
  float* qkv_f  = (float*)alloc((size_t)T_ * QKVW_ * 4);
  bf16* q_bf    = (bf16*)alloc((size_t)NH_ * T_ * D_ * 2);
  bf16* k_bf    = (bf16*)alloc((size_t)NKV_ * T_ * D_ * 2);
  bf16* vT_bf   = (bf16*)alloc((size_t)NKV_ * D_ * T_ * 2);
  float* scores = (float*)alloc((size_t)NH_ * T_ * T_ * 4);
  bf16* probs   = (bf16*)alloc((size_t)NH_ * T_ * T_ * 2);
  float* o_f    = (float*)alloc((size_t)T_ * H_ * 4);
  bf16* o_bf    = (bf16*)alloc((size_t)T_ * H_ * 2);
  float* attn_f = (float*)alloc((size_t)T_ * H_ * 4);
  bf16* h2_bf   = (bf16*)alloc((size_t)T_ * H_ * 2);
  float* logits = (float*)alloc((size_t)T_ * E_ * 4);
  int*   cnt    = (int*)  alloc((size_t)E_ * 4);
  int*   tok    = (int*)  alloc((size_t)E_ * T_ * 4);
  float* slotw  = (float*)alloc((size_t)E_ * T_ * 4);
  bf16* act     = (bf16*)alloc((size_t)E_ * T_ * I_ * 2);

  // ---- zero accumulators ----
  hipMemsetAsync(moe_out, 0, (size_t)T_ * H_ * 4, stream);
  hipMemsetAsync(cnt, 0, E_ * 4, stream);

  // ---- weight conversion (streams ~204 MB once @ 23.3 TB/s) ----
  cvt_bf16<<<2048, 256, 0, stream>>>(qkv_w, qkvw_bf, QKVW_ * H_);
  cvt_bf16<<<2048, 256, 0, stream>>>(o_w, ow_bf, H_ * H_);
  cvt_bf16<<<128, 256, 0, stream>>>(gate_w, gw_bf, E_ * H_);
  cvt_t_bf16<<<dim3(512, 1, E_), 256, 0, stream>>>(w_gate, wg_t, H_, I_); // -> (E,I,H)
  cvt_t_bf16<<<dim3(512, 1, E_), 256, 0, stream>>>(w_up,   wu_t, H_, I_); // -> (E,I,H)
  cvt_t_bf16<<<dim3(512, 1, E_), 256, 0, stream>>>(w_down, wd_t, I_, H_); // -> (E,H,I)

  // ---- 1. residual add + RMSNorm(ln1) ----
  add_rmsnorm<<<T_, 256, 0, stream>>>(hidden, residual, ln1_w, res_out, h_bf);

  // ---- 2. QKV projection: (1024x1024) x (2048x1024)^T ----
  gemm_nt<<<dim3(QKVW_ / BN_, T_ / BM_), 128, 0, stream>>>(
      h_bf, qkvw_bf, qkv_f, T_, QKVW_, H_, H_, H_, QKVW_, 0);

  // ---- 3. q/k norm + RoPE + v transpose ----
  qk_rope<<<T_, 128, 0, stream>>>(qkv_f, positions, q_norm_w, k_norm_w,
                                  q_bf, k_bf, vT_bf);

  // ---- 4. attention scores per head (causal block skip) ----
  for (int h = 0; h < NH_; ++h) {
    gemm_nt<<<dim3(T_ / BN_, T_ / BM_), 128, 0, stream>>>(
        q_bf + (size_t)h * T_ * D_, k_bf + (size_t)(h / 2) * T_ * D_,
        scores + (size_t)h * T_ * T_, T_, T_, D_, D_, D_, T_, 1);
  }

  // ---- 5. causal softmax -> bf16 probs ----
  softmax_causal<<<dim3(T_, NH_), 256, 0, stream>>>(scores, probs);

  // ---- 6. P*V per head (K clamped near diagonal), output [T][NH*D] ----
  for (int h = 0; h < NH_; ++h) {
    gemm_nt<<<dim3(D_ / BN_, T_ / BM_), 128, 0, stream>>>(
        probs + (size_t)h * T_ * T_, vT_bf + (size_t)(h / 2) * D_ * T_,
        o_f + (size_t)h * D_, T_, D_, T_, T_, T_, NH_ * D_, 2);
  }
  cvt_bf16<<<2048, 256, 0, stream>>>(o_f, o_bf, T_ * H_);

  // ---- 7. output projection ----
  gemm_nt<<<dim3(H_ / BN_, T_ / BM_), 128, 0, stream>>>(
      o_bf, ow_bf, attn_f, T_, H_, H_, H_, H_, H_, 0);

  // ---- 8. residual add + RMSNorm(ln2) ----
  add_rmsnorm<<<T_, 256, 0, stream>>>(attn_f, res_out, ln2_w, res_out, h2_bf);

  // ---- 9. router logits + top-4 scatter ----
  gemm_nt<<<dim3(1, T_ / BM_), 128, 0, stream>>>(
      h2_bf, gw_bf, logits, T_, E_, H_, H_, H_, E_, 0);
  router_topk<<<T_, 32, 0, stream>>>(logits, cnt, tok, slotw);

  // ---- 10. sparse MoE ----
  moe_gate_up<<<dim3(I_ / 64, T_ / 16, E_), 128, 0, stream>>>(
      h2_bf, wg_t, wu_t, cnt, tok, act);
  moe_down<<<dim3(H_ / 64, T_ / 16, E_), 128, 0, stream>>>(
      act, wd_t, cnt, tok, slotw, moe_out);
}
